// PABDMH_metapath_specific_48833778155892
// MI455X (gfx1250) — compile-verified, matrix operand order checked by reading, not run on hardware
//
#include <hip/hip_runtime.h>

typedef __attribute__((ext_vector_type(16))) _Float16 v16h;
typedef __attribute__((ext_vector_type(8)))  float    v8f;

#define ALPHA 0.001f

union Frag16 { v16h v; uint4 u[2]; };
union H8 { uint4 u; _Float16 h[8]; };

#if defined(__HIP_DEVICE_COMPILE__) && \
    __has_builtin(__builtin_amdgcn_global_load_async_to_lds_b128)
#define HAS_ASYNC_LDS 1
#else
#define HAS_ASYNC_LDS 0
#endif

// K index held by (lane, half-slot i) in a 16-bit 16x32 A/B WMMA fragment
// (CDNA5 ISA 7.12.2: VGPR0..3 -> K=0..7 (+8 for lanes 16..31),
//  VGPR4..7 -> K=16..23 (+8 for lanes 16..31))
__device__ __forceinline__ int kmap_idx(int lane, int i) {
  int hi = (lane >> 4) & 1;
  int v = i >> 1, e = i & 1;
  return (v < 4) ? (2 * v + e + 8 * hi) : (16 + 2 * (v - 4) + e + 8 * hi);
}

// order-preserving float <-> uint for atomicMax-based segment max
__device__ __forceinline__ unsigned f2o(float f) {
  unsigned u = __float_as_uint(f);
  return (u & 0x80000000u) ? ~u : (u | 0x80000000u);
}
__device__ __forceinline__ float o2f(unsigned u) {
  return (u & 0x80000000u) ? __uint_as_float(u ^ 0x80000000u)
                           : __uint_as_float(~u);
}

__device__ __forceinline__ v8f splat8(float x) {
  v8f r;
#pragma unroll
  for (int i = 0; i < 8; ++i) r[i] = x;
  return r;
}

__device__ __forceinline__ v8f wmma_f16(v16h a, v16h b, v8f c) {
  return __builtin_amdgcn_wmma_f32_16x16x32_f16(
      /*neg_a=*/false, a, /*neg_b=*/false, b,
      /*c_mod=*/(short)0, c, /*reuse_a=*/false, /*reuse_b=*/false);
}

// pre-swizzled B fragment: 32 lanes x 16 halves, lane-major, 1KB per fragment
__device__ __forceinline__ v16h load_frag(const _Float16* base, int fragIdx,
                                          int lane) {
  return *(const v16h*)(base + (long)fragIdx * 512 + lane * 16);
}

#if HAS_ASYNC_LDS
// builtin expects int4 pointers (vector_size(16) int) in AS1 / AS3
typedef int v4i_ __attribute__((vector_size(16)));
typedef __attribute__((address_space(1))) v4i_ as1_v4i;
typedef __attribute__((address_space(3))) v4i_ as3_v4i;

template <int N>
__device__ __forceinline__ void wait_asynccnt() {
#if __has_builtin(__builtin_amdgcn_s_wait_asynccnt)
  __builtin_amdgcn_s_wait_asynccnt(N);
#else
  asm volatile("s_wait_asynccnt %0" ::"n"(N));
#endif
  asm volatile("" ::: "memory");  // keep LDS reads below the wait
}

// copy one 1KB B fragment (32 lanes x 32B) global -> LDS, 2 async b128/lane
__device__ __forceinline__ void async_stage_frag(const _Float16* gfrag,
                                                 _Float16* lfrag, int lane) {
  _Float16* g = const_cast<_Float16*>(gfrag) + lane * 16;
  _Float16* l = lfrag + lane * 16;
  __builtin_amdgcn_global_load_async_to_lds_b128((as1_v4i*)g, (as3_v4i*)l, 0,
                                                 0);
  __builtin_amdgcn_global_load_async_to_lds_b128((as1_v4i*)(g + 8),
                                                 (as3_v4i*)(l + 8), 0, 0);
}
#endif

// ---------------------------------------------------------------------------
// Pack w_ih [1536,64] and w_hh [1536,512] (f32, row-major (j,k)) into f16
// WMMA-B fragments. Fragment (ncol, kb) covers B[k, n] = w[ncol*16+n][kb*32+k'].
// w_ih: 96 cols * 2 kb; w_hh: 96 cols * 16 kb. 1KB per fragment, lane-major.
// ---------------------------------------------------------------------------
__global__ void __launch_bounds__(32)
pack_weights_kernel(const float* __restrict__ w_ih,
                    const float* __restrict__ w_hh,
                    _Float16* __restrict__ wih_f,
                    _Float16* __restrict__ whh_f) {
  int b = blockIdx.x, lane = threadIdx.x;
  const float* src;
  _Float16* dstp;
  int ncol, kb, K;
  if (b < 192) {
    ncol = b >> 1; kb = b & 1; K = 64;
    src = w_ih; dstp = wih_f + (long)b * 512;
  } else {
    int b2 = b - 192;
    ncol = b2 >> 4; kb = b2 & 15; K = 512;
    src = w_hh; dstp = whh_f + (long)b2 * 512;
  }
  int n_ = lane & 15;
  Frag16 f;
  _Float16* fh = (_Float16*)&f;
#pragma unroll
  for (int i = 0; i < 16; ++i) {
    int k = kb * 32 + kmap_idx(lane, i);
    fh[i] = (_Float16)src[(long)(ncol * 16 + n_) * K + k];
  }
  *(v16h*)(dstp + lane * 16) = f.v;
}

// ---------------------------------------------------------------------------
// One wave per 32-edge tile (2 WMMA M-tiles): each B fragment feeds both
// M-tiles. h state double-buffered in dynamic LDS (f16). w_hh B fragments are
// prefetched one k-step ahead into LDS via GLOBAL_LOAD_ASYNC_TO_LDS_B128
// (ASYNCcnt-tracked), then consumed with ds_load_b128 -> v_wmma.
// ---------------------------------------------------------------------------
__global__ void __launch_bounds__(32)
gru_attn_kernel(const float* __restrict__ features,
                const int* __restrict__ emi,
                const int* __restrict__ dst,
                const float* __restrict__ b_ih,
                const float* __restrict__ b_hh,
                const float* __restrict__ attn,
                const _Float16* __restrict__ wih_f,
                const _Float16* __restrict__ whh_f,
                float* __restrict__ hidden,
                float* __restrict__ a_out,
                unsigned* __restrict__ segmax,
                int E, int L) {
  extern __shared__ __align__(16) char smem_raw[];
  _Float16* smem = (_Float16*)smem_raw;  // h: [buf:2][tile:2][row:16][520]
  _Float16* bstage = smem + 2 * 2 * 16 * 520;  // [buf:2][gate:3][512]

  const int lane = threadIdx.x;
  const int hi = lane >> 4;
  const int lp = lane & 15;
  const int tile0 = blockIdx.x * 32;

  auto hrow = [&](int buf, int t, int row) -> _Float16* {
    return smem + (long)(((buf * 2 + t) * 16 + row)) * 520;
  };

  // zero buffer 0 (h_prev for step 0), both tiles
  {
    uint4 z = {0u, 0u, 0u, 0u};
    uint4* p = (uint4*)smem;
    for (int i = lane; i < (2 * 16 * 520 * 2) / 16; i += 32) p[i] = z;
  }
  __syncthreads();

  v16h xfrag[2][2];

#pragma unroll 1
  for (int step = 0; step < L; ++step) {
    const int rb = step & 1, wb = rb ^ 1;

    // Build x A-fragments directly from global feature rows (L2-resident):
    // fragment = two 8-half runs: [kb*32+8*hi, +8) and [kb*32+16+8*hi, +8)
#pragma unroll
    for (int t = 0; t < 2; ++t) {
      int e = tile0 + t * 16 + lp;
      if (e >= E) e = E - 1;
      int node = emi[(long)e * L + step];
      const float* rowp = features + (long)node * 64;
#pragma unroll
      for (int kb = 0; kb < 2; ++kb) {
        const float* rp = rowp + kb * 32 + hi * 8;
        float4 f0 = *(const float4*)(rp);
        float4 f1 = *(const float4*)(rp + 4);
        float4 f2 = *(const float4*)(rp + 16);
        float4 f3 = *(const float4*)(rp + 20);
        Frag16 f;
        _Float16* fh = (_Float16*)&f;
        fh[0] = (_Float16)f0.x; fh[1] = (_Float16)f0.y;
        fh[2] = (_Float16)f0.z; fh[3] = (_Float16)f0.w;
        fh[4] = (_Float16)f1.x; fh[5] = (_Float16)f1.y;
        fh[6] = (_Float16)f1.z; fh[7] = (_Float16)f1.w;
        fh[8]  = (_Float16)f2.x; fh[9]  = (_Float16)f2.y;
        fh[10] = (_Float16)f2.z; fh[11] = (_Float16)f2.w;
        fh[12] = (_Float16)f3.x; fh[13] = (_Float16)f3.y;
        fh[14] = (_Float16)f3.z; fh[15] = (_Float16)f3.w;
        xfrag[t][kb] = f.v;
      }
    }

    // 32 N-chunks of 16 columns (gate column-blocks: r=c, z=32+c, n=64+c)
#pragma unroll 1
    for (int c = 0; c < 32; ++c) {
      int col = c * 16 + lp;
      v8f aR[2], aZ[2], aN[2], aH[2];
      {
        float bR = b_ih[col] + b_hh[col];
        float bZ = b_ih[512 + col] + b_hh[512 + col];
        float bN = b_ih[1024 + col];
        float bH = b_hh[1024 + col];
#pragma unroll
        for (int t = 0; t < 2; ++t) {
          aR[t] = splat8(bR); aZ[t] = splat8(bZ);
          aN[t] = splat8(bN); aH[t] = splat8(bH);
        }
      }

      // input-to-hidden part (K = 64)
#pragma unroll
      for (int kb = 0; kb < 2; ++kb) {
        v16h bR = load_frag(wih_f, (c) * 2 + kb, lane);
        aR[0] = wmma_f16(xfrag[0][kb], bR, aR[0]);
        aR[1] = wmma_f16(xfrag[1][kb], bR, aR[1]);
        v16h bZ = load_frag(wih_f, (32 + c) * 2 + kb, lane);
        aZ[0] = wmma_f16(xfrag[0][kb], bZ, aZ[0]);
        aZ[1] = wmma_f16(xfrag[1][kb], bZ, aZ[1]);
        v16h bN = load_frag(wih_f, (64 + c) * 2 + kb, lane);
        aN[0] = wmma_f16(xfrag[0][kb], bN, aN[0]);
        aN[1] = wmma_f16(xfrag[1][kb], bN, aN[1]);
      }

      // hidden-to-hidden part (K = 512); h0 == 0 so step 0 is skipped
      if (step > 0) {
        const _Float16* r0 = hrow(rb, 0, lp);
        const _Float16* r1 = hrow(rb, 1, lp);

#if HAS_ASYNC_LDS
        // stage kb=0 fragments (3 gates) into LDS buffer 0
        {
          _Float16* sb = bstage;
          async_stage_frag(whh_f + ((long)(c) * 16 + 0) * 512, sb, lane);
          async_stage_frag(whh_f + ((long)(32 + c) * 16 + 0) * 512, sb + 512, lane);
          async_stage_frag(whh_f + ((long)(64 + c) * 16 + 0) * 512, sb + 1024, lane);
        }
#pragma unroll
        for (int kb = 0; kb < 16; ++kb) {
          if (kb < 15) {  // stage kb+1 into the other buffer
            _Float16* sb = bstage + ((kb + 1) & 1) * 1536;
            async_stage_frag(whh_f + ((long)(c) * 16 + kb + 1) * 512, sb, lane);
            async_stage_frag(whh_f + ((long)(32 + c) * 16 + kb + 1) * 512,
                             sb + 512, lane);
            async_stage_frag(whh_f + ((long)(64 + c) * 16 + kb + 1) * 512,
                             sb + 1024, lane);
            wait_asynccnt<6>();  // kb's 6 async loads retired, kb+1 in flight
          } else {
            wait_asynccnt<0>();
          }
          const _Float16* sb = bstage + (kb & 1) * 1536;
          Frag16 fa0, fa1;
          fa0.u[0] = *(const uint4*)(r0 + kb * 32 + hi * 8);
          fa0.u[1] = *(const uint4*)(r0 + kb * 32 + 16 + hi * 8);
          fa1.u[0] = *(const uint4*)(r1 + kb * 32 + hi * 8);
          fa1.u[1] = *(const uint4*)(r1 + kb * 32 + 16 + hi * 8);
          v16h bR = *(const v16h*)(sb + lane * 16);
          aR[0] = wmma_f16(fa0.v, bR, aR[0]);
          aR[1] = wmma_f16(fa1.v, bR, aR[1]);
          v16h bZ = *(const v16h*)(sb + 512 + lane * 16);
          aZ[0] = wmma_f16(fa0.v, bZ, aZ[0]);
          aZ[1] = wmma_f16(fa1.v, bZ, aZ[1]);
          v16h bN = *(const v16h*)(sb + 1024 + lane * 16);
          aH[0] = wmma_f16(fa0.v, bN, aH[0]);
          aH[1] = wmma_f16(fa1.v, bN, aH[1]);
        }
#else
        if (c + 1 < 32) {  // fallback: prefetch next chunk's B streams
          __builtin_prefetch(whh_f + (long)(c + 1) * 16 * 512, 0, 1);
          __builtin_prefetch(whh_f + (long)(32 + c + 1) * 16 * 512, 0, 1);
          __builtin_prefetch(whh_f + (long)(64 + c + 1) * 16 * 512, 0, 1);
        }
#pragma unroll
        for (int kb = 0; kb < 16; ++kb) {
          Frag16 fa0, fa1;
          fa0.u[0] = *(const uint4*)(r0 + kb * 32 + hi * 8);
          fa0.u[1] = *(const uint4*)(r0 + kb * 32 + 16 + hi * 8);
          fa1.u[0] = *(const uint4*)(r1 + kb * 32 + hi * 8);
          fa1.u[1] = *(const uint4*)(r1 + kb * 32 + 16 + hi * 8);
          v16h bR = load_frag(whh_f, (c) * 16 + kb, lane);
          aR[0] = wmma_f16(fa0.v, bR, aR[0]);
          aR[1] = wmma_f16(fa1.v, bR, aR[1]);
          v16h bZ = load_frag(whh_f, (32 + c) * 16 + kb, lane);
          aZ[0] = wmma_f16(fa0.v, bZ, aZ[0]);
          aZ[1] = wmma_f16(fa1.v, bZ, aZ[1]);
          v16h bN = load_frag(whh_f, (64 + c) * 16 + kb, lane);
          aH[0] = wmma_f16(fa0.v, bN, aH[0]);
          aH[1] = wmma_f16(fa1.v, bN, aH[1]);
        }
#endif
      }

      // GRU update on this 16x16 chunk per tile; C layout: m = v + 8*hi.
      // Reads h_prev from buffer rb, writes h_new to buffer wb (no hazard).
#pragma unroll
      for (int t = 0; t < 2; ++t) {
#pragma unroll
        for (int v = 0; v < 8; ++v) {
          int m = v + 8 * hi;
          float hp = (float)hrow(rb, t, m)[col];
          float r_ = 1.0f / (1.0f + __expf(-aR[t][v]));
          float z_ = 1.0f / (1.0f + __expf(-aZ[t][v]));
          float nn = tanhf(aN[t][v] + r_ * aH[t][v]);
          hrow(wb, t, m)[col] = (_Float16)((1.0f - z_) * nn + z_ * hp);
        }
      }
    }
    __syncthreads();
  }

  const int fin = L & 1;  // buffer holding the final hidden state

  // spill hidden [32,512] f32 to global
#pragma unroll
  for (int t = 0; t < 2; ++t) {
    for (int q = lane; q < 1024; q += 32) {
      int row = q >> 6, cc = q & 63;
      int e = tile0 + t * 16 + row;
      if (e < E) {
        H8 x;
        x.u = *(const uint4*)(hrow(fin, t, row) + cc * 8);
        float4 o0 = {(float)x.h[0], (float)x.h[1], (float)x.h[2], (float)x.h[3]};
        float4 o1 = {(float)x.h[4], (float)x.h[5], (float)x.h[6], (float)x.h[7]};
        float* p = hidden + (long)e * 512 + cc * 8;
        *(float4*)p = o0;
        *(float4*)(p + 4) = o1;
      }
    }
  }

  // attention logits: a[e,h] = leakyrelu(sum_d h[e, h*64+d] * attn[h,d])
#pragma unroll
  for (int t = 0; t < 2; ++t) {
#pragma unroll
    for (int j = 0; j < 4; ++j) {
      int p = j * 32 + lane;
      int er = p & 15, hh2 = p >> 4;
      const _Float16* hr = hrow(fin, t, er);
      float s = 0.f;
      for (int d = 0; d < 64; ++d)
        s += (float)hr[hh2 * 64 + d] * attn[hh2 * 64 + d];
      float av = s > 0.f ? s : ALPHA * s;
      int e = tile0 + t * 16 + er;
      if (e < E) {
        a_out[(long)e * 8 + hh2] = av;
        atomicMax(&segmax[(long)dst[e] * 8 + hh2], f2o(av));
      }
    }
  }
}

// exp(a - segmax[dst]) in place + denominator accumulation
__global__ void expsum_kernel(float* __restrict__ a_buf,
                              const int* __restrict__ dst,
                              const unsigned* __restrict__ segmax,
                              float* __restrict__ denom, int E) {
  int i = blockIdx.x * blockDim.x + threadIdx.x;
  if (i >= E * 8) return;
  int e = i >> 3, hh = i & 7;
  float amax = o2f(segmax[(long)dst[e] * 8 + hh]);
  float ex = __expf(a_buf[i] - amax);
  a_buf[i] = ex;
  atomicAdd(&denom[(long)dst[e] * 8 + hh], ex);
}

// ft[dst, h, d] += hidden[e, h*64+d] * a_exp[e,h] / denom[dst,h]
__global__ void scatter_kernel(const float* __restrict__ hidden,
                               const float* __restrict__ a_ex,
                               const float* __restrict__ denom,
                               const int* __restrict__ dst,
                               float* __restrict__ out, int E) {
  long i = (long)blockIdx.x * blockDim.x + threadIdx.x;
  if (i >= (long)E * 512) return;
  int e = (int)(i >> 9), r = (int)(i & 511), hh = r >> 6;
  int dn = dst[e];
  float att = a_ex[(long)e * 8 + hh] / fmaxf(denom[(long)dn * 8 + hh], 1e-12f);
  atomicAdd(out + (long)dn * 512 + r, hidden[i] * att);
}

extern "C" void kernel_launch(void* const* d_in, const int* in_sizes, int n_in,
                              void* d_out, int out_size, void* d_ws,
                              size_t ws_size, hipStream_t stream) {
  const float* features = (const float*)d_in[0];
  const int* emi = (const int*)d_in[1];
  const int* dst = (const int*)d_in[2];
  // d_in[3] = num_nodes scalar (derived from out_size instead)
  const float* w_ih = (const float*)d_in[4];
  const float* w_hh = (const float*)d_in[5];
  const float* b_ih = (const float*)d_in[6];
  const float* b_hh = (const float*)d_in[7];
  const float* attn = (const float*)d_in[8];

  int E = in_sizes[2];
  int L = in_sizes[1] / E;
  int N = out_size / 512;  // H*D = 512

  // workspace layout
  char* ws = (char*)d_ws;
  _Float16* wih_f = (_Float16*)ws;             // 192 frags * 1KB
  _Float16* whh_f = (_Float16*)(ws + 196608);  // 1536 frags * 1KB
  size_t off = 196608 + 1572864;               // = 1769472
  float* hidden = (float*)(ws + off);
  off += (size_t)E * 512 * 4;
  float* a_buf = (float*)(ws + off);
  off += (size_t)E * 8 * 4;
  unsigned* segmax = (unsigned*)(ws + off);
  off += (size_t)N * 8 * 4;
  float* denom = (float*)(ws + off);

  // zero output + softmax state (segmax/denom adjacent)
  (void)hipMemsetAsync(d_out, 0, (size_t)out_size * 4, stream);
  (void)hipMemsetAsync(segmax, 0, (size_t)N * 8 * 4 * 2, stream);

  pack_weights_kernel<<<192 + 1536, 32, 0, stream>>>(w_ih, w_hh, wih_f, whh_f);

  // dynamic LDS: h (2 x 2 x 16 x 520 halves = 66,560 B) + B stage (6,144 B)
  const int smem_bytes = 2 * 2 * 16 * 520 * 2 + 2 * 3 * 512 * 2;
  (void)hipFuncSetAttribute((const void*)gru_attn_kernel,
                            hipFuncAttributeMaxDynamicSharedMemorySize,
                            smem_bytes);

  int tiles32 = (E + 31) / 32;
  gru_attn_kernel<<<tiles32, 32, smem_bytes, stream>>>(
      features, emi, dst, b_ih, b_hh, attn, wih_f, whh_f, hidden, a_buf,
      segmax, E, L);

  int eh = E * 8;
  expsum_kernel<<<(eh + 255) / 256, 256, 0, stream>>>(a_buf, dst, segmax,
                                                      denom, E);

  long tot = (long)E * 512;
  scatter_kernel<<<(int)((tot + 255) / 256), 256, 0, stream>>>(
      hidden, a_buf, denom, dst, (float*)d_out, E);
}
